// RIMSEncoder_62148176773399
// MI455X (gfx1250) — matrix operand order repaired and checked by reading, not compiled, not run on hardware
//
#include <hip/hip_runtime.h>
#include <hip/hip_bf16.h>

// ---------------------------------------------------------------------------
// RIMS encoder, fused for MI455X (gfx1250, wave32, WMMA bf16 16x16x32).
// dims: R=16 rims, C=64, HD=4, D=16, K=4 ; B=8, H=W=64 -> 32768 pixels.
// Each wave owns TWO 16-pixel tiles (32 pixels): every weight A-fragment is
// loaded once and multiplied against two independent B-fragment sets, halving
// weight traffic and doubling WMMA ILP (hides the 16-bit WMMA RAW hazard).
// All four GEMMs (key, val, merge, out-proj) run on v_wmma_f32_16x16x32_bf16
// with f32 accumulation; the zero "null" batch collapses to constants and the
// 2-way softmax to a sigmoid, so the 134MB keyf/val tensors never exist.
// ---------------------------------------------------------------------------

#define R_    16
#define C_    64
#define HD_   4
#define D_    16
#define TOPK_ 4
#define B_    8
#define HW_   4096      // 64*64
#define OC_   1024      // R*HD*D
#define NPIX_ 32768     // B*HW

typedef __bf16 bf16_t;
typedef bf16_t v16bf __attribute__((ext_vector_type(16)));
typedef float  v8f   __attribute__((ext_vector_type(8)));

__device__ __forceinline__ v8f wmma_bf16(v16bf a, v16bf b, v8f c) {
  // D = A(16x32) x B(32x16) + C, f32 accumulate
  return __builtin_amdgcn_wmma_f32_16x16x32_bf16(
      /*neg_a=*/false, a, /*neg_b=*/false, b,
      /*c_mod=*/(short)0, c, /*reuse_a=*/false, /*reuse_b=*/false);
}

// --- workspace layout (bytes) ---------------------------------------------
#define WS_WKA 0         // bf16 A-frags of Wk  [64 mt][2 kt][32 lanes][16]
#define WS_WVA 131072    // bf16 A-frags of Wv
#define WS_WMA 262144    // bf16 A-frags of Wm (flat [1024,64])
#define WS_WOA 393216    // bf16 A-frags of Wo  [4][2][32][16]
#define WS_Q   401408    // f32 [1024] relu(rims.Wq + bq)
#define WS_IPN 405504    // f32 [64]   null-key inner products per (rim,head)
#define WS_IDX 405760    // u32 [32768] packed top-4 nibbles per pixel

// ---------------------------------------------------------------------------
// Pack a row-major f32 [mtiles*16, 64] matrix into bf16 WMMA A-fragments.
// A-fragment order used here (B matches): lane l holds row M = l%16; element e of the v16bf
// maps to K = e + 16*(l>=16)  (contiguous, not the ISA table order).
// ---------------------------------------------------------------------------
__global__ __launch_bounds__(256) void pack_afrag(const float* __restrict__ src,
                                                  bf16_t* __restrict__ dst,
                                                  int mtiles) {
  int t = blockIdx.x * 256 + threadIdx.x;
  int total = mtiles * 2 * 32 * 16;
  if (t >= total) return;
  int e    = t & 15;
  int lane = (t >> 4) & 31;
  int kt   = (t >> 9) & 1;
  int mt   = t >> 10;
  int row  = mt * 16 + (lane & 15);
  int k    = kt * 32 + e + ((lane >= 16) ? 16 : 0);
  dst[t] = (bf16_t)src[row * 64 + k];
}

// q[o] = relu( dot(rims[r,:], Wq[o,:]) + bq[o] ),  r = o/64
__global__ __launch_bounds__(256) void compute_q(const float* __restrict__ rims,
                                                 const float* __restrict__ Wq,
                                                 const float* __restrict__ bq,
                                                 float* __restrict__ q) {
  int o = blockIdx.x * 256 + threadIdx.x;
  if (o >= OC_) return;
  int r = o >> 6;
  float s = bq[o];
  #pragma unroll 8
  for (int c = 0; c < C_; ++c) s += rims[r * C_ + c] * Wq[o * C_ + c];
  q[o] = fmaxf(s, 0.0f);
}

// ip_null[t] = sum_d relu(bk[t*16+d]) * q[t*16+d]   (t indexes (rim,head))
__global__ __launch_bounds__(64) void compute_ipnull(const float* __restrict__ bk,
                                                     const float* __restrict__ q,
                                                     float* __restrict__ ipn) {
  int t = threadIdx.x;
  if (t >= R_ * HD_) return;
  float s = 0.0f;
  #pragma unroll
  for (int d = 0; d < D_; ++d) {
    int o = t * 16 + d;
    s += fmaxf(bk[o], 0.0f) * q[o];
  }
  ipn[t] = s;
}

// ---------------------------------------------------------------------------
// Kernel 1: per wave, two 16-pixel tiles. scores[r] = sum_h sigmoid(ip-ipn),
// then top-4 per pixel -> packed u32. 8 waves / block, 128 blocks.
// ---------------------------------------------------------------------------
__global__ __launch_bounds__(256) void scores_topk(const float* __restrict__ x,
                                                   const bf16_t* __restrict__ WkA,
                                                   const float* __restrict__ bk,
                                                   const float* __restrict__ q,
                                                   const float* __restrict__ ipn,
                                                   unsigned* __restrict__ idx_out) {
  __shared__ float sc_lds[8][R_][32];          // [wave][rim][2 tiles x 16 px]
  const int lane = threadIdx.x & 31;
  const int wv   = threadIdx.x >> 5;
  const int tp   = blockIdx.x * 8 + wv;        // tile-pair index
  const int pl   = lane & 15;
  const int hi   = lane >> 4;

  int pixv[2];
  v16bf xb[2][2];                              // [tile][ktile]
  #pragma unroll
  for (int t = 0; t < 2; ++t) {
    const int pix = tp * 32 + t * 16 + pl;
    pixv[t] = pix;
    const int b  = pix >> 12;
    const int hw = pix & (HW_ - 1);
    #pragma unroll
    for (int kt = 0; kt < 2; ++kt)
      #pragma unroll
      for (int e = 0; e < 16; ++e) {
        int c = kt * 32 + e + 16 * hi;
        xb[t][kt][e] = (bf16_t)x[(b * C_ + c) * HW_ + hw];
      }
  }

  for (int r = 0; r < R_; ++r) {
    float sr[2] = {0.0f, 0.0f};
    #pragma unroll
    for (int hh = 0; hh < HD_; ++hh) {
      const int mt = r * 4 + hh;               // 16 output channels = one (r,h)
      const v16bf a0 = *(const v16bf*)(WkA + ((mt * 2 + 0) * 32 + lane) * 16);
      const v16bf a1 = *(const v16bf*)(WkA + ((mt * 2 + 1) * 32 + lane) * 16);
      // two independent accumulator chains hide the WMMA RAW hazard
      v8f acc0 = {}, acc1 = {};
      acc0 = wmma_bf16(a0, xb[0][0], acc0);
      acc1 = wmma_bf16(a0, xb[1][0], acc1);
      acc0 = wmma_bf16(a1, xb[0][1], acc0);
      acc1 = wmma_bf16(a1, xb[1][1], acc1);
      const int o8 = mt * 16 + 8 * hi;         // C/D layout: VGPR j -> ch o8+j
      float bk8[8], q8[8];
      #pragma unroll
      for (int j = 0; j < 8; ++j) { bk8[j] = bk[o8 + j]; q8[j] = q[o8 + j]; }
      const float ipnull = ipn[mt];
      #pragma unroll
      for (int t = 0; t < 2; ++t) {
        float s = 0.0f;
        #pragma unroll
        for (int j = 0; j < 8; ++j) {
          float kf = fmaxf((t ? acc1[j] : acc0[j]) + bk8[j], 0.0f);
          s += kf * q8[j];
        }
        s += __shfl_xor(s, 16, 32);            // join the two channel halves
        sr[t] += 1.0f / (1.0f + __expf(-(s - ipnull)));
      }
    }
    if (hi == 0) {
      sc_lds[wv][r][pl]      = sr[0];
      sc_lds[wv][r][16 + pl] = sr[1];
    }
  }

  // Top-4 per pixel (strict > keeps lowest index on ties, like jax top_k).
  #pragma unroll
  for (int t = 0; t < 2; ++t) {
    float sv[16];
    #pragma unroll
    for (int i = 0; i < 16; ++i) sv[i] = sc_lds[wv][i][t * 16 + pl];
    unsigned packed = 0;
    #pragma unroll
    for (int k = 0; k < TOPK_; ++k) {
      int best = 0;
      float bvv = sv[0];
      #pragma unroll
      for (int i = 1; i < 16; ++i)
        if (sv[i] > bvv) { bvv = sv[i]; best = i; }
      packed |= (unsigned)best << (4 * k);
      #pragma unroll
      for (int i = 0; i < 16; ++i)
        if (i == best) sv[i] = -1e30f;
    }
    if (hi == 0) idx_out[pixv[t]] = packed;
  }
}

// ---------------------------------------------------------------------------
// Kernel 2: per wave, two 16-pixel tiles. For each rim: val GEMM -> LDS
// re-layout -> merge GEMM (+bm in C operand) -> masked 0.25 accumulate; then
// relu -> LDS re-layout -> out-projection GEMM (+bo), relu, store.
// ---------------------------------------------------------------------------
__global__ __launch_bounds__(256) void merge_out(const float* __restrict__ x,
                                                 const bf16_t* __restrict__ WvA,
                                                 const float* __restrict__ bv,
                                                 const bf16_t* __restrict__ WmA,
                                                 const float* __restrict__ bm,
                                                 const bf16_t* __restrict__ WoA,
                                                 const float* __restrict__ bo,
                                                 const unsigned* __restrict__ idx_in,
                                                 float* __restrict__ out) {
  __shared__ float vlds[8][2][C_ * 16];        // wave-private [tile][ch][px]
  const int lane = threadIdx.x & 31;
  const int wv   = threadIdx.x >> 5;
  const int tp   = blockIdx.x * 8 + wv;
  const int pl   = lane & 15;
  const int hi   = lane >> 4;

  int bv_[2], hwv[2];
  unsigned selv[2];
  v16bf xb[2][2];
  #pragma unroll
  for (int t = 0; t < 2; ++t) {
    const int pix = tp * 32 + t * 16 + pl;
    bv_[t] = pix >> 12;
    hwv[t] = pix & (HW_ - 1);
    #pragma unroll
    for (int kt = 0; kt < 2; ++kt)
      #pragma unroll
      for (int e = 0; e < 16; ++e) {
        int c = kt * 32 + e + 16 * hi;
        xb[t][kt][e] = (bf16_t)x[(bv_[t] * C_ + c) * HW_ + hwv[t]];
      }
    const unsigned pk = idx_in[pix];
    unsigned sel = 0;
    #pragma unroll
    for (int k = 0; k < TOPK_; ++k) sel |= 1u << ((pk >> (4 * k)) & 15u);
    selv[t] = sel;
  }

  v8f oacc[2][4] = {{{}, {}, {}, {}}, {{}, {}, {}, {}}};

  for (int r = 0; r < R_; ++r) {
    // --- GEMM1: val_r[64 x 16px] for both tiles, shared A-fragments ---
    #pragma unroll
    for (int ct = 0; ct < 4; ++ct) {
      const int mt = r * 4 + ct;
      const v16bf a0 = *(const v16bf*)(WvA + ((mt * 2 + 0) * 32 + lane) * 16);
      const v16bf a1 = *(const v16bf*)(WvA + ((mt * 2 + 1) * 32 + lane) * 16);
      const int o8 = mt * 16 + 8 * hi;
      v8f acc0 = {}, acc1 = {};
      acc0 = wmma_bf16(a0, xb[0][0], acc0);
      acc1 = wmma_bf16(a0, xb[1][0], acc1);
      acc0 = wmma_bf16(a1, xb[0][1], acc0);
      acc1 = wmma_bf16(a1, xb[1][1], acc1);
      #pragma unroll
      for (int j = 0; j < 8; ++j) {
        const int ch = ct * 16 + j + 8 * hi;   // channel within rim (0..63)
        const float bvj = bv[o8 + j];
        vlds[wv][0][ch * 16 + pl] = fmaxf(acc0[j] + bvj, 0.0f);
        vlds[wv][1][ch * 16 + pl] = fmaxf(acc1[j] + bvj, 0.0f);
      }
    }
    // --- re-layout: f32 accumulators -> bf16 B-fragments (K = val channel) ---
    v16bf vb[2][2];
    #pragma unroll
    for (int t = 0; t < 2; ++t)
      #pragma unroll
      for (int kt = 0; kt < 2; ++kt)
        #pragma unroll
        for (int e = 0; e < 16; ++e) {
          int ch = kt * 32 + e + 16 * hi;
          vb[t][kt][e] = (bf16_t)vlds[wv][t][ch * 16 + pl];
        }
    // --- GEMM2: merged_r = Wm_r . val_r + bm_r, masked mean accumulate ---
    const float m0 = ((selv[0] >> r) & 1u) ? 0.25f : 0.0f;
    const float m1 = ((selv[1] >> r) & 1u) ? 0.25f : 0.0f;
    #pragma unroll
    for (int ct = 0; ct < 4; ++ct) {
      const int mt = r * 4 + ct;
      const v16bf a0 = *(const v16bf*)(WmA + ((mt * 2 + 0) * 32 + lane) * 16);
      const v16bf a1 = *(const v16bf*)(WmA + ((mt * 2 + 1) * 32 + lane) * 16);
      const int o8 = mt * 16 + 8 * hi;         // bm is R*C flat
      v8f mc0, mc1;
      #pragma unroll
      for (int j = 0; j < 8; ++j) { mc0[j] = bm[o8 + j]; mc1[j] = mc0[j]; }
      mc0 = wmma_bf16(a0, vb[0][0], mc0);
      mc1 = wmma_bf16(a0, vb[1][0], mc1);
      mc0 = wmma_bf16(a1, vb[0][1], mc0);
      mc1 = wmma_bf16(a1, vb[1][1], mc1);
      #pragma unroll
      for (int j = 0; j < 8; ++j) {
        oacc[0][ct][j] += m0 * mc0[j];
        oacc[1][ct][j] += m1 * mc1[j];
      }
    }
  }

  // relu(mean) -> LDS -> B-fragments for the output projection.
  #pragma unroll
  for (int t = 0; t < 2; ++t)
    #pragma unroll
    for (int ct = 0; ct < 4; ++ct)
      #pragma unroll
      for (int j = 0; j < 8; ++j) {
        int ch = ct * 16 + j + 8 * hi;
        vlds[wv][t][ch * 16 + pl] = fmaxf(oacc[t][ct][j], 0.0f);
      }
  v16bf ob[2][2];
  #pragma unroll
  for (int t = 0; t < 2; ++t)
    #pragma unroll
    for (int kt = 0; kt < 2; ++kt)
      #pragma unroll
      for (int e = 0; e < 16; ++e) {
        int ch = kt * 32 + e + 16 * hi;
        ob[t][kt][e] = (bf16_t)vlds[wv][t][ch * 16 + pl];
      }

  // --- GEMM3: out = relu(Wo . (...) + bo), shared A-fragments ---
  #pragma unroll
  for (int ct = 0; ct < 4; ++ct) {
    const v16bf a0 = *(const v16bf*)(WoA + ((ct * 2 + 0) * 32 + lane) * 16);
    const v16bf a1 = *(const v16bf*)(WoA + ((ct * 2 + 1) * 32 + lane) * 16);
    v8f f0, f1;
    #pragma unroll
    for (int j = 0; j < 8; ++j) { f0[j] = bo[ct * 16 + j + 8 * hi]; f1[j] = f0[j]; }
    f0 = wmma_bf16(a0, ob[0][0], f0);
    f1 = wmma_bf16(a0, ob[1][0], f1);
    f0 = wmma_bf16(a1, ob[0][1], f0);
    f1 = wmma_bf16(a1, ob[1][1], f1);
    #pragma unroll
    for (int j = 0; j < 8; ++j) {
      const int c = ct * 16 + j + 8 * hi;
      out[(bv_[0] * C_ + c) * HW_ + hwv[0]] = fmaxf(f0[j], 0.0f);
      out[(bv_[1] * C_ + c) * HW_ + hwv[1]] = fmaxf(f1[j], 0.0f);
    }
  }
}

// ---------------------------------------------------------------------------
extern "C" void kernel_launch(void* const* d_in, const int* in_sizes, int n_in,
                              void* d_out, int out_size, void* d_ws, size_t ws_size,
                              hipStream_t stream) {
  const float* x    = (const float*)d_in[0];
  const float* rims = (const float*)d_in[1];
  const float* Wk   = (const float*)d_in[2];
  const float* bk   = (const float*)d_in[3];
  const float* Wv   = (const float*)d_in[4];
  const float* bv   = (const float*)d_in[5];
  const float* Wq   = (const float*)d_in[6];
  const float* bq   = (const float*)d_in[7];
  const float* Wm   = (const float*)d_in[8];
  const float* bm   = (const float*)d_in[9];
  const float* Wo   = (const float*)d_in[10];
  const float* bo   = (const float*)d_in[11];
  float* out = (float*)d_out;

  char* ws = (char*)d_ws;
  bf16_t*   WkA = (bf16_t*)(ws + WS_WKA);
  bf16_t*   WvA = (bf16_t*)(ws + WS_WVA);
  bf16_t*   WmA = (bf16_t*)(ws + WS_WMA);
  bf16_t*   WoA = (bf16_t*)(ws + WS_WOA);
  float*    qb  = (float*)(ws + WS_Q);
  float*    ipn = (float*)(ws + WS_IPN);
  unsigned* idx = (unsigned*)(ws + WS_IDX);

  // Weight re-layout to bf16 WMMA A-fragments.
  pack_afrag<<<(64 * 1024 + 255) / 256, 256, 0, stream>>>(Wk, WkA, 64);
  pack_afrag<<<(64 * 1024 + 255) / 256, 256, 0, stream>>>(Wv, WvA, 64);
  pack_afrag<<<(64 * 1024 + 255) / 256, 256, 0, stream>>>(Wm, WmA, 64);
  pack_afrag<<<(4 * 1024 + 255) / 256, 256, 0, stream>>>(Wo, WoA, 4);

  compute_q<<<(OC_ + 255) / 256, 256, 0, stream>>>(rims, Wq, bq, qb);
  compute_ipnull<<<1, 64, 0, stream>>>(bk, qb, ipn);

  // 32768 pixels / (8 waves * 32 pixels) = 128 blocks.
  scores_topk<<<NPIX_ / 256, 256, 0, stream>>>(x, WkA, bk, qb, ipn, idx);
  merge_out<<<NPIX_ / 256, 256, 0, stream>>>(x, WvA, bv, WmA, bm, WoA, bo, idx, out);
}
